// UniG_24670292148525
// MI455X (gfx1250) — compile-verified, matrix operand order checked by reading, not run on hardware
//
#include <hip/hip_runtime.h>
#include <hip/hip_bf16.h>
#include <stdint.h>

// ---------------- CDNA5 WMMA vector types ----------------
typedef __attribute__((ext_vector_type(16))) __bf16 v16bf;
typedef __attribute__((ext_vector_type(8)))  __bf16 v8bf;
typedef __attribute__((ext_vector_type(8)))  float  v8f;

union BF16x16 { v16bf v; v8bf h[2]; };

// Async global->LDS copy path (gfx1250). Guarded so the file compiles on
// toolchains without the builtins (falls back to global-load + ds_store).
#if defined(__AMDGCN__) && __has_builtin(__builtin_amdgcn_global_load_async_to_lds_b128) && \
    __has_builtin(__builtin_amdgcn_s_wait_asynccnt)
#define USE_ASYNC_LDS 1
// Builtin param types per hipcc diagnostic: v4i (gcc vector_size(16)) pointers.
typedef int v4i_vs __attribute__((vector_size(16)));
typedef v4i_vs __attribute__((address_space(1)))* async_gptr_t;  // global src
typedef v4i_vs __attribute__((address_space(3)))* async_lptr_t;  // LDS dst (32-bit)
#else
#define USE_ASYNC_LDS 0
#endif

// ---------------- utility kernels ----------------
__global__ void k_zero_f32(float* __restrict__ p, long long n) {
  long long i = (long long)blockIdx.x * blockDim.x + threadIdx.x;
  long long stride = (long long)gridDim.x * blockDim.x;
  for (; i < n; i += stride) p[i] = 0.0f;
}

__global__ void k_f32_to_bf16(const float* __restrict__ in, __bf16* __restrict__ out,
                              long long n) {
  long long i = (long long)blockIdx.x * blockDim.x + threadIdx.x;
  long long stride = (long long)gridDim.x * blockDim.x;
  for (; i < n; i += stride) out[i] = (__bf16)in[i];
}

__global__ void k_pad_bias(const float* __restrict__ b, float* __restrict__ out,
                           int n, int npad) {
  int i = blockIdx.x * blockDim.x + threadIdx.x;
  if (i < npad) out[i] = (i < n) ? b[i] : 0.0f;
}

// ---------------- SpMM: h[row] += val * x[col]  (fp32 atomic scatter) ----------------
__global__ void k_spmm_scatter_f32(const float* __restrict__ x,
                                   const float* __restrict__ vals,
                                   const int* __restrict__ rows,
                                   const int* __restrict__ cols,
                                   float* __restrict__ h, int nnz) {
  long long t = (long long)blockIdx.x * blockDim.x + threadIdx.x;
  long long total = (long long)nnz * 128;
  if (t >= total) return;
  int e = (int)(t >> 7);
  int q = (int)(t & 127);
  float v = vals[e];
  int src = cols[e];
  int dst = rows[e];
  const float4 xv = *(const float4*)(x + (size_t)src * 512 + q * 4);
  float* hp = h + (size_t)dst * 512 + q * 4;
  atomicAdd(hp + 0, v * xv.x);
  atomicAdd(hp + 1, v * xv.y);
  atomicAdd(hp + 2, v * xv.z);
  atomicAdd(hp + 3, v * xv.w);
}

// ---------------- WMMA GEMM with LDS-staged B ----------------
// out = act(A[M,K] @ W[NOUT,K]^T + bias). Per block (8 waves): the NT*16 x 32
// B-slab is staged into LDS once (async copy when available, double-buffered)
// and shared by all waves; each wave then runs NT WMMAs per K-step with B
// fragments from ds_load_b128. A fragments stream from global per wave.
template <int K, int NT, bool RELU, bool OUTBF>
__global__ __launch_bounds__(256) void k_gemm_wmma(const __bf16* __restrict__ A,
                                                   const __bf16* __restrict__ W,
                                                   const float* __restrict__ bias,
                                                   void* __restrict__ outv,
                                                   int M, int LDO) {
  constexpr int ROWS   = NT * 16;     // B rows (output cols) per block
  constexpr int NCHUNK = ROWS * 4;    // 16-byte chunks per 32-wide K slab
  constexpr int NS     = K / 32;      // number of K steps
  __shared__ __align__(16) __bf16 Bs[2][ROWS * 32];

  int tid  = threadIdx.x;
  int lane = tid & 31;
  int wave = tid >> 5;
  int m0   = (blockIdx.x * 8 + wave) * 16;
  bool active = (m0 < M);             // no early return: all waves join barriers
  int mrow = active ? m0 : 0;         // clamp A address for inactive waves
  int n0   = blockIdx.y * ROWS;
  int r    = lane & 15;               // row within A-tile / col within B-tile
  int hi   = lane >> 4;               // K-half selector per ISA 16-bit layout

  const __bf16* wtile = W + (size_t)n0 * K;             // [ROWS][K]
  const __bf16* arow  = A + (size_t)(mrow + r) * K;

  // Cooperative copy of one 32-wide B slab (k0..k0+31) into Bs[buf].
  auto stage = [&](int buf, int k0) {
    for (int c = tid; c < NCHUNK; c += 256) {
      int row  = c >> 2;
      int koff = (c & 3) * 8;
      const __bf16* src = wtile + (size_t)row * K + k0 + koff;
      __bf16* dst = &Bs[buf][row * 32 + koff];
#if USE_ASYNC_LDS
      // Integer round-trips avoid cross-AS pointer-cast restrictions; generic
      // shared addresses carry the LDS offset in addr[31:0] (ISA aperture map).
      __builtin_amdgcn_global_load_async_to_lds_b128(
          (async_gptr_t)(uintptr_t)src,
          (async_lptr_t)(uint32_t)(uintptr_t)dst, 0, 0);
#else
      *(v8bf*)dst = *(const v8bf*)src;
#endif
    }
  };

  v8f acc[NT] = {};

  stage(0, 0);
#if USE_ASYNC_LDS
  __builtin_amdgcn_s_wait_asynccnt(0);
#endif
  __syncthreads();

  for (int ks = 0; ks < NS; ++ks) {
    int buf = ks & 1;
    if (ks + 1 < NS) stage(buf ^ 1, (ks + 1) * 32);  // prefetch next slab

    int k0 = ks * 32;
    BF16x16 a;
    a.h[0] = *(const v8bf*)(arow + k0 + hi * 8);        // K = k0+hi*8 .. +7
    a.h[1] = *(const v8bf*)(arow + k0 + 16 + hi * 8);   // K = k0+16+hi*8 .. +7
#pragma unroll
    for (int t = 0; t < NT; ++t) {
      BF16x16 b;
      b.h[0] = *(const v8bf*)&Bs[buf][(t * 16 + r) * 32 + hi * 8];
      b.h[1] = *(const v8bf*)&Bs[buf][(t * 16 + r) * 32 + 16 + hi * 8];
      acc[t] = __builtin_amdgcn_wmma_f32_16x16x32_bf16(
          /*neg_a=*/false, a.v, /*neg_b=*/false, b.v,
          /*c_mod=*/(short)0, acc[t], /*reuse_a=*/false, /*reuse_b=*/false);
    }

#if USE_ASYNC_LDS
    __builtin_amdgcn_s_wait_asynccnt(0);
#endif
    __syncthreads();   // next slab visible; previous buffer safe to overwrite
  }

  if (!active) return;
  // C/D layout: VGPR g, lanes 0-15 -> M = g, lanes 16-31 -> M = 8+g; N = lane&15.
#pragma unroll
  for (int t = 0; t < NT; ++t) {
    int n = n0 + t * 16 + r;
    float bn = bias[n];
#pragma unroll
    for (int g = 0; g < 8; ++g) {
      int m = m0 + g + 8 * hi;
      float val = acc[t][g] + bn;
      if (RELU) val = fmaxf(val, 0.0f);
      if (OUTBF) ((__bf16*)outv)[(size_t)m * LDO + n] = (__bf16)val;
      else       ((float*)outv)[(size_t)m * LDO + n] = val;
    }
  }
}

// ---------------- transpose SpMM: y[col] += val * h[row], h padded [N,48] ----------------
__global__ void k_spmm_t_scatter(const float* __restrict__ vals,
                                 const int* __restrict__ rows,
                                 const int* __restrict__ cols,
                                 const float* __restrict__ h,   // [N,48] f32
                                 float* __restrict__ y,         // [N,40] f32
                                 int nnz) {
  long long t = (long long)blockIdx.x * blockDim.x + threadIdx.x;
  long long total = (long long)nnz * 40;
  if (t >= total) return;
  int e = (int)(t / 40);
  int j = (int)(t - (long long)e * 40);
  float v = vals[e];
  int src = rows[e];
  int dst = cols[e];
  atomicAdd(y + (size_t)dst * 40 + j, v * h[(size_t)src * 48 + j]);
}

// ---------------- row-wise log_softmax over 40 classes ----------------
__global__ void k_log_softmax40(const float* __restrict__ y, float* __restrict__ out, int n) {
  int rIdx = blockIdx.x * blockDim.x + threadIdx.x;
  if (rIdx >= n) return;
  const float* p = y + (size_t)rIdx * 40;
  float m = -3.4e38f;
  for (int j = 0; j < 40; ++j) m = fmaxf(m, p[j]);
  float s = 0.0f;
  for (int j = 0; j < 40; ++j) s += __expf(p[j] - m);
  float ls = __logf(s);
  float* o = out + (size_t)rIdx * 40;
  for (int j = 0; j < 40; ++j) o[j] = (p[j] - m) - ls;
}

// ---------------- host launcher ----------------
extern "C" void kernel_launch(void* const* d_in, const int* in_sizes, int n_in,
                              void* d_out, int out_size, void* d_ws, size_t ws_size,
                              hipStream_t stream) {
  const float* x       = (const float*)d_in[0];
  const float* pv_vals = (const float*)d_in[1];
  const float* W0      = (const float*)d_in[2];
  const float* b0      = (const float*)d_in[3];
  const float* W1      = (const float*)d_in[4];
  const float* b1      = (const float*)d_in[5];
  const float* W2      = (const float*)d_in[6];
  const float* b2      = (const float*)d_in[7];
  // d_in[8] = adj (unused)
  const int* pv_rows   = (const int*)d_in[9];
  const int* pv_cols   = (const int*)d_in[10];

  const int NFEAT = 512, NHID = 256, NCLASS = 40, NCP = 48;
  const int N   = in_sizes[0] / NFEAT;
  const int NNZ = in_sizes[1];

  char* ws = (char*)d_ws;
  size_t off = 0;
  auto alloc = [&](size_t bytes) {
    size_t o = off;
    off += (bytes + 255) & ~(size_t)255;
    return o;
  };
  float*  h0f  = (float*)(ws + alloc((size_t)N * NFEAT * 4));  // fp32 SpMM accumulator
  __bf16* h0b  = (__bf16*)(ws + alloc((size_t)N * NFEAT * 2)); // bf16 acts (reused for h2)
  __bf16* h1b  = (__bf16*)(ws + alloc((size_t)N * NHID * 2));
  float*  h2f  = (float*)(ws + alloc((size_t)N * NCP * 4));    // logits, padded to 48
  float*  yacc = (float*)(ws + alloc((size_t)N * NCLASS * 4));
  __bf16* W0b  = (__bf16*)(ws + alloc((size_t)NHID * NFEAT * 2));
  __bf16* W1b  = (__bf16*)(ws + alloc((size_t)NHID * NHID * 2));
  __bf16* W2b  = (__bf16*)(ws + alloc((size_t)NCP * NHID * 2));
  float*  b2p  = (float*)(ws + alloc((size_t)NCP * 4));
  (void)ws_size; (void)n_in; (void)out_size;

  const long long nh0 = (long long)N * NFEAT;

  // zero accumulators + padded weight buffer
  k_zero_f32<<<4096, 256, 0, stream>>>(h0f, nh0);
  k_zero_f32<<<1024, 256, 0, stream>>>(yacc, (long long)N * NCLASS);
  k_zero_f32<<<32, 256, 0, stream>>>((float*)W2b, (long long)NCP * NHID / 2);

  // weights -> bf16 (W2 written into zero-padded 48x256 buffer)
  k_f32_to_bf16<<<512, 256, 0, stream>>>(W0, W0b, (long long)NHID * NFEAT);
  k_f32_to_bf16<<<256, 256, 0, stream>>>(W1, W1b, (long long)NHID * NHID);
  k_f32_to_bf16<<<64, 256, 0, stream>>>(W2, W2b, (long long)NCLASS * NHID);
  k_pad_bias<<<1, 64, 0, stream>>>(b2, b2p, NCLASS, NCP);

  // h0 = Pv @ x (fp32 atomic scatter)
  long long tot1 = (long long)NNZ * (NFEAT / 4);
  k_spmm_scatter_f32<<<(int)((tot1 + 255) / 256), 256, 0, stream>>>(
      x, pv_vals, pv_rows, pv_cols, h0f, NNZ);

  // cast activations to bf16 for the matrix cores
  k_f32_to_bf16<<<8192, 256, 0, stream>>>(h0f, h0b, nh0);

  // dense stack via v_wmma_f32_16x16x32_bf16 with LDS-staged weights
  int gblocks = (N + 127) / 128;  // 8 waves/block, 16 rows/wave
  dim3 g2(gblocks, 2);            // y-dim spans the two 128-col groups of NHID=256
  k_gemm_wmma<512, 8, true,  true ><<<g2, 256, 0, stream>>>(h0b, W0b, b0,  h1b, N, NHID);
  k_gemm_wmma<256, 8, true,  true ><<<g2, 256, 0, stream>>>(h1b, W1b, b1,  h0b, N, NHID);
  k_gemm_wmma<256, 3, false, false><<<dim3(gblocks, 1), 256, 0, stream>>>(
      h0b, W2b, b2p, h2f, N, NCP);

  // y = PvT @ logits (fp32 atomic scatter)
  long long tot2 = (long long)NNZ * NCLASS;
  k_spmm_t_scatter<<<(int)((tot2 + 255) / 256), 256, 0, stream>>>(
      pv_vals, pv_rows, pv_cols, h2f, yacc, NNZ);

  // log_softmax -> d_out
  k_log_softmax40<<<(N + 255) / 256, 256, 0, stream>>>(yacc, (float*)d_out, N);
}
